// Poly2d_50087908606313
// MI455X (gfx1250) — matrix operand order repaired and checked by reading, not compiled, not run on hardware
//
#include <hip/hip_runtime.h>
#include <hip/hip_bf16.h>

typedef __attribute__((ext_vector_type(16))) _Float16 v16h;
typedef __attribute__((ext_vector_type(8)))  _Float16 v8h;
typedef __attribute__((ext_vector_type(8)))  float    v8f;

// Problem: x (16,64,32,32) f32, filters (64,64,10,10) f32 -> out (16,64,32,32) f32
// out[b,o,y,x] = sum_{c,m,n} f[o,c,m,n] * s_m * s_n,  s = [1, 3x3 patch] (10 vals)
// Symmetric fold: m<=n pairs only (55), filters pre-folded; K padded 55 -> 64.
#define PB 16
#define PC 64
#define PO 64
#define PH 32
#define PW 32
#define KSYM 55
#define KPAD 64

// Triangular pair tables: kk -> (m,n), m<=n, row-major over m. constexpr so
// unrolled accesses fold to literals (keeps s[] in registers).
__device__ constexpr int TRI_M[KSYM] = {
    0,0,0,0,0,0,0,0,0,0,  1,1,1,1,1,1,1,1,1,  2,2,2,2,2,2,2,2,
    3,3,3,3,3,3,3,        4,4,4,4,4,4,        5,5,5,5,5,
    6,6,6,6,              7,7,7,              8,8,  9 };
__device__ constexpr int TRI_N[KSYM] = {
    0,1,2,3,4,5,6,7,8,9,  1,2,3,4,5,6,7,8,9,  2,3,4,5,6,7,8,9,
    3,4,5,6,7,8,9,        4,5,6,7,8,9,        5,6,7,8,9,
    6,7,8,9,              7,8,9,              8,9,  9 };

__device__ __host__ inline void tri_mn_rt(int idx, int& m, int& n) {
    int mm = 0, rem = idx;
    while (rem >= 10 - mm) { rem -= 10 - mm; ++mm; }
    m = mm; n = mm + rem;
}

// ---------------------------------------------------------------------------
// Kernel 1: fold + repack filters -> Bf[c][o][kk] f16, kk in [0,64).
// ---------------------------------------------------------------------------
__global__ void poly2d_pack_filters(const float* __restrict__ F,
                                    _Float16* __restrict__ Bf) {
    int idx = blockIdx.x * blockDim.x + threadIdx.x;   // 64*64*64 = 262144
    if (idx >= PC * PO * KPAD) return;
    int kk = idx & (KPAD - 1);
    int o  = (idx >> 6) & 63;
    int c  = idx >> 12;
    float v = 0.0f;
    if (kk < KSYM) {
        int m, n; tri_mn_rt(kk, m, n);
        const float* fb = F + (o * PC + c) * 100;
        v = fb[m * 10 + n];
        if (m != n) v += fb[n * 10 + m];
    }
    Bf[idx] = (_Float16)v;
}

// Compute 16 consecutive feature halves (kk = KK0..KK0+15) for one pixel from
// its 10 spot values; fully unrolled, (m,n) fold to compile-time constants.
template <int KK0>
__device__ inline void feat16(const float s[10], _Float16* __restrict__ dst) {
    v8h lo, hhi;
    #pragma unroll
    for (int i = 0; i < 8; ++i) {
        const int kk = KK0 + i;
        float v = 0.0f;
        if (kk < KSYM) v = s[TRI_M[kk]] * s[TRI_N[kk]];
        lo[i] = (_Float16)v;
    }
    #pragma unroll
    for (int i = 0; i < 8; ++i) {
        const int kk = KK0 + 8 + i;
        float v = 0.0f;
        if (kk < KSYM) v = s[TRI_M[kk]] * s[TRI_N[kk]];
        hhi[i] = (_Float16)v;
    }
    *(v8h*)dst       = lo;   // 16B-aligned (32B-aligned half offset)
    *(v8h*)(dst + 8) = hhi;
}

// ---------------------------------------------------------------------------
// Kernel 2: fused feat-generation + WMMA GEMM.
// Block = 128 threads (4 waves), one image row (b,y): 32 pixels x 64 outputs.
// Feat stage: lane = pixel, wave = 16-wide kk group; spot read directly from
// global (L0-resident 4KB plane), products in registers, 2x b128 LDS stores.
// GEMM stage: wave w owns cols [16w,16w+16), B in regs, 2 M-subtiles.
// Single barrier per channel; ldsA double-buffered.
// ---------------------------------------------------------------------------
__global__ void __launch_bounds__(128)
poly2d_main(const float* __restrict__ x,
            const _Float16* __restrict__ Bf,
            float* __restrict__ out) {
    __shared__ _Float16 ldsA[2][32 * KPAD];   // 2 x 4 KiB A tiles

    const int tid = threadIdx.x;
    const int g   = blockIdx.x;            // 0..511 -> (b, y)
    const int b   = g >> 5;
    const int y   = g & 31;

    const int wave = tid >> 5;
    const int lane = tid & 31;
    const int hi   = lane >> 4;            // 0: lanes 0-15, 1: lanes 16-31
    const int row  = lane & 15;            // A M-row / B column N

    // Spot gather constants: lane = pixel p. Clamped offsets + validity flags
    // (branchless loads; clamped address always in-plane).
    const int p = lane;
    int  offs[9];
    bool ok[9];
    #pragma unroll
    for (int j = 0; j < 9; ++j) {
        const int dy = j / 3, dx = j - dy * 3;
        const int yy = y + dy - 1;
        const int xx = p + dx - 1;
        ok[j]   = (yy >= 0) & (yy < PH) & (xx >= 0) & (xx < PW);
        const int yc = yy < 0 ? 0 : (yy > PH - 1 ? PH - 1 : yy);
        const int xc = xx < 0 ? 0 : (xx > PW - 1 ? PW - 1 : xx);
        offs[j] = (yc << 5) + xc;
    }
    const float* xplane = x + ((b * PC) << 10);

    // Stage-3 per-lane LDS offsets (contiguous k order, same as B):
    //   lanes 0-15 : halves K{ks*32+0..15}
    //   lanes 16-31: halves K{ks*32+16..31}
    const int aoff0 = (row << 6)        + (hi ? 16 : 0);   // pixels 0-15
    const int aoff1 = ((16 + row) << 6) + (hi ? 16 : 0);   // pixels 16-31

    v8f acc0 = {};
    v8f acc1 = {};

    for (int c = 0; c < PC; ++c) {
        _Float16* abuf = &ldsA[c & 1][0];

        // ---- B loads first: in flight across feat compute + barrier ----
        const _Float16* bbase =
            Bf + ((((c << 6) + (wave << 4) + row) << 6) + (hi ? 16 : 0));
        v16h b0 = *(const v16h*)(bbase);
        v16h b1 = *(const v16h*)(bbase + 32);

        // ---- Spot gather (registers, branchless) ----
        const float* xc = xplane + (c << 10);
        float s[10];
        s[0] = 1.0f;
        #pragma unroll
        for (int j = 0; j < 9; ++j) {
            float v = xc[offs[j]];
            s[1 + j] = ok[j] ? v : 0.0f;
        }
        // WGP-scope prefetch of next channel's halo line (all cache levels).
        if (c + 1 < PC)
            __builtin_prefetch(xc + (1 << 10) + offs[4], 0, 3);

        // ---- Feat products -> LDS (wave-uniform kk group) ----
        _Float16* dst = abuf + (p << 6) + (wave << 4);
        switch (wave) {
            case 0: feat16<0>(s, dst);  break;
            case 1: feat16<16>(s, dst); break;
            case 2: feat16<32>(s, dst); break;
            default: feat16<48>(s, dst); break;
        }
        __syncthreads();

        // ---- WMMA: 2 K-steps x 2 M-subtiles (all A loads batched first) ----
        const _Float16* pa00 = abuf + aoff0;
        const _Float16* pa10 = abuf + aoff1;
        v8h l00 = *(const v8h*)pa00,        h00 = *(const v8h*)(pa00 + 8);
        v8h l10 = *(const v8h*)pa10,        h10 = *(const v8h*)(pa10 + 8);
        v8h l01 = *(const v8h*)(pa00 + 32), h01 = *(const v8h*)(pa00 + 40);
        v8h l11 = *(const v8h*)(pa10 + 32), h11 = *(const v8h*)(pa10 + 40);
        v16h a00 = __builtin_shufflevector(l00, h00, 0,1,2,3,4,5,6,7,8,9,10,11,12,13,14,15);
        v16h a10 = __builtin_shufflevector(l10, h10, 0,1,2,3,4,5,6,7,8,9,10,11,12,13,14,15);
        v16h a01 = __builtin_shufflevector(l01, h01, 0,1,2,3,4,5,6,7,8,9,10,11,12,13,14,15);
        v16h a11 = __builtin_shufflevector(l11, h11, 0,1,2,3,4,5,6,7,8,9,10,11,12,13,14,15);

        acc0 = __builtin_amdgcn_wmma_f32_16x16x32_f16(false, a00, false, b0, (short)0, acc0, false, false);
        acc1 = __builtin_amdgcn_wmma_f32_16x16x32_f16(false, a10, false, b0, (short)0, acc1, false, false);
        acc0 = __builtin_amdgcn_wmma_f32_16x16x32_f16(false, a01, false, b1, (short)0, acc0, false, false);
        acc1 = __builtin_amdgcn_wmma_f32_16x16x32_f16(false, a11, false, b1, (short)0, acc1, false, false);
        // No trailing barrier: next channel writes the other ldsA buffer, and
        // this buffer's reads are complete (dscnt waited before WMMA) before
        // any wave signals the next channel's barrier.
    }

    // ---- Epilogue: C layout -> out[b][o][y][x] ----
    // lanes 0-15: VGPR r -> M=r; lanes 16-31: M = 8+r. N = lane & 15.
    const int o = (wave << 4) + row;
    float* orow = out + (((b << 6) + o) << 10) + (y << 5);
    #pragma unroll
    for (int r = 0; r < 8; ++r) {
        orow[(hi << 3) + r]      = acc0[r];   // pixels 0-15
        orow[16 + (hi << 3) + r] = acc1[r];   // pixels 16-31
    }
}

extern "C" void kernel_launch(void* const* d_in, const int* in_sizes, int n_in,
                              void* d_out, int out_size, void* d_ws, size_t ws_size,
                              hipStream_t stream) {
    const float* x = (const float*)d_in[0];    // (16,64,32,32) f32
    const float* F = (const float*)d_in[1];    // (64,64,10,10) f32
    float* out     = (float*)d_out;            // (16,64,32,32) f32
    _Float16* Bf   = (_Float16*)d_ws;          // 64*64*64 halves = 512 KiB

    {   // fold + repack filters
        const int total = PC * PO * KPAD;      // 262144
        const int blk   = 256;
        poly2d_pack_filters<<<(total + blk - 1) / blk, blk, 0, stream>>>(F, Bf);
    }
    {   // fused feat + WMMA GEMM: 512 blocks (one image row each) x 128 threads
        poly2d_main<<<512, 128, 0, stream>>>(x, Bf, out);
    }
}
